// Seq2Seq_300647711198
// MI455X (gfx1250) — compile-verified, matrix operand order checked by reading, not run on hardware
//
#include <hip/hip_runtime.h>
#include <hip/hip_bf16.h>
#include <math.h>

// ---------------- problem constants (match reference) ----------------
#define S_   64      // max_length / encoder steps
#define T_   32      // decoder target length (uses T_-1 steps)
#define B_   512     // batch
#define IN_  66      // encoder input features
#define INP_ 96      // IN padded up to multiple of 32 for WMMA K
#define H_   1024    // hidden
#define O_   1024    // output
#define H3_  3072

typedef __bf16 bf16_t;
typedef __attribute__((ext_vector_type(16))) __bf16 v16bf;
typedef __attribute__((ext_vector_type(8)))  float  v8f;

// ---------------------------------------------------------------------
// One K-segment of the WMMA GEMM. Wave computes a 32x64 C tile:
// 2 A-fragments x 4 B-fragments -> 8 v_wmma per 32-wide K slab.
// Fragment layouts used here (contiguous k order, same for A and B):
//   A (16x32 bf16): lane l holds row (l&15), K-half (l>>4)*16 -> one
//     contiguous 32B load from row-major A.
//   B (32x16 bf16): identical per-lane pattern from row-major W [N,K]
//     (C = A * W^T makes W rows the B columns).
// ---------------------------------------------------------------------
__device__ __forceinline__ void gemm_segment(
    const bf16_t* __restrict__ A, int lda,
    const bf16_t* __restrict__ W, int ldw, int K,
    int m0, int n0, int rsel, int khalf, v8f (&acc)[2][4])
{
    const bf16_t* a0p = A + (size_t)(m0 + rsel) * lda + khalf;
    const bf16_t* a1p = a0p + (size_t)16 * lda;
    const bf16_t* wp  = W + (size_t)(n0 + rsel) * ldw + khalf;
    for (int k = 0; k < K; k += 32) {
        v16bf a0 = *(const v16bf*)(a0p + k);
        v16bf a1 = *(const v16bf*)(a1p + k);
#pragma unroll
        for (int j = 0; j < 4; ++j) {
            v16bf w = *(const v16bf*)(wp + (size_t)j * 16 * ldw + k);
            acc[0][j] = __builtin_amdgcn_wmma_f32_16x16x32_bf16(
                false, a0, false, w, (short)0, acc[0][j], false, false);
            acc[1][j] = __builtin_amdgcn_wmma_f32_16x16x32_bf16(
                false, a1, false, w, (short)0, acc[1][j], false, false);
        }
    }
}

// ---------------------------------------------------------------------
// C[M,N] = A0[M,K0]*W0[N,K0]^T (+ A1[M,K1]*W1[N,K1]^T) (+bias) (+ReLU)
// The optional second segment implements [x|h] @ W^T without any
// concatenation pass (W1 = W + K0 within rows of stride ldw).
// Block = 256 threads = 8 waves arranged (8/NW) m-rows x NW n-cols;
// each wave owns a 32x64 tile. Grid: x = N/(64*NW), y = M/(32*(8/NW)).
// ---------------------------------------------------------------------
template<int NW>
__global__ __launch_bounds__(256)
void gemm_bf16_wmma(const bf16_t* __restrict__ A0, int lda0,
                    const bf16_t* __restrict__ W0, int ldw0, int K0,
                    const bf16_t* __restrict__ A1, int lda1,
                    const bf16_t* __restrict__ W1, int ldw1, int K1,
                    const float*  __restrict__ bias,
                    float*        __restrict__ Cf,   // nullable
                    bf16_t*       __restrict__ Cb,   // nullable
                    int N, int act)
{
    constexpr int MW = 8 / NW;
    const int lane  = threadIdx.x & 31;
    const int wave  = threadIdx.x >> 5;
    const int m0    = blockIdx.y * (32 * MW) + (wave / NW) * 32;
    const int n0    = blockIdx.x * (64 * NW) + (wave % NW) * 64;
    const int rsel  = lane & 15;          // A row / W row (=C col) within tile
    const int khalf = (lane >> 4) * 16;   // lanes 16..31 hold K=16..31 of a slab

    v8f acc[2][4];
#pragma unroll
    for (int mi = 0; mi < 2; ++mi)
#pragma unroll
        for (int j = 0; j < 4; ++j)
            acc[mi][j] = (v8f){0.f,0.f,0.f,0.f,0.f,0.f,0.f,0.f};

    gemm_segment(A0, lda0, W0, ldw0, K0, m0, n0, rsel, khalf, acc);
    if (A1) gemm_segment(A1, lda1, W1, ldw1, K1, m0, n0, rsel, khalf, acc);

    // C layout: VGPR i -> row m0 + mi*16 + (lane>>4)*8 + i,
    //           col n0 + j*16 + (lane&15)
#pragma unroll
    for (int mi = 0; mi < 2; ++mi) {
#pragma unroll
        for (int j = 0; j < 4; ++j) {
            const int c   = n0 + j * 16 + rsel;
            const float bb = bias ? bias[c] : 0.0f;
#pragma unroll
            for (int i = 0; i < 8; ++i) {
                const int r = m0 + mi * 16 + (lane >> 4) * 8 + i;
                float v = acc[mi][j][i] + bb;
                if (act) v = fmaxf(v, 0.0f);
                const size_t o = (size_t)r * N + c;
                if (Cf) Cf[o] = v;
                if (Cb) Cb[o] = (bf16_t)v;
            }
        }
    }
}

// ------------- f32 -> bf16 convert with optional column padding -------------
__global__ void cvt_pad_bf16(const float* __restrict__ src, bf16_t* __restrict__ dst,
                             long long rows, int sc, int dc)
{
    long long idx = (long long)blockIdx.x * blockDim.x + threadIdx.x;
    if (idx >= rows * dc) return;
    long long r = idx / dc; int c = (int)(idx % dc);
    dst[idx] = (c < sc) ? (bf16_t)src[r * sc + c] : (bf16_t)0.0f;
}

// ------------- zero init hidden state (f32 + bf16 copies) -------------
__global__ void init_h(float* h, bf16_t* hb, int n)
{
    int i = blockIdx.x * blockDim.x + threadIdx.x;
    if (i < n) { h[i] = 0.0f; hb[i] = (bf16_t)0.0f; }
}

// ------------- GRU pointwise: gates ordered (r,z,n), PyTorch style -------------
// Writes h (f32, recurrence), hb (bf16, feeds next-step WMMA GEMMs) and
// optionally the bf16 encoder-output slice consumed by the attention context.
__global__ void gru_pointwise(const float* __restrict__ gi, const float* __restrict__ gh,
                              float* __restrict__ h, bf16_t* __restrict__ hb,
                              bf16_t* __restrict__ enc_slice_b /*nullable*/)
{
    int idx = blockIdx.x * blockDim.x + threadIdx.x;   // b*H + j
    if (idx >= B_ * H_) return;
    int b = idx >> 10, j = idx & (H_ - 1);
    size_t base = (size_t)b * H3_;
    float ir = gi[base + j], iz = gi[base + H_ + j], inn = gi[base + 2 * H_ + j];
    float hr = gh[base + j], hz = gh[base + H_ + j], hn  = gh[base + 2 * H_ + j];
    float r = 1.0f / (1.0f + __expf(-(ir + hr)));
    float z = 1.0f / (1.0f + __expf(-(iz + hz)));
    float n = tanhf(inn + r * hn);
    float h2 = (1.0f - z) * n + z * h[idx];
    h[idx]  = h2;
    bf16_t hv = (bf16_t)h2;
    hb[idx] = hv;
    if (enc_slice_b) enc_slice_b[idx] = hv;
}

// ------------- softmax over 64 attention logits; 1 block (64 thr) per row -------------
__global__ void softmax64(const float* __restrict__ in, float* __restrict__ out)
{
    __shared__ float red[64];
    int b = blockIdx.x, t = threadIdx.x;
    float v = in[(size_t)b * S_ + t];
    red[t] = v; __syncthreads();
    for (int s = 32; s > 0; s >>= 1) { if (t < s) red[t] = fmaxf(red[t], red[t + s]); __syncthreads(); }
    float mx = red[0]; __syncthreads();
    float e = __expf(v - mx);
    red[t] = e; __syncthreads();
    for (int s = 32; s > 0; s >>= 1) { if (t < s) red[t] += red[t + s]; __syncthreads(); }
    out[(size_t)b * S_ + t] = e / red[0];
}

// ------------- ctx[b,h] = sum_l aw[b,l] * enc_out[l,b,h] (bf16 stream, f32 acc) ----
__global__ void ctx_einsum(const float* __restrict__ aw, const bf16_t* __restrict__ enc_out_b,
                           bf16_t* __restrict__ ctx_b)
{
    int idx = blockIdx.x * blockDim.x + threadIdx.x;   // b*H + h
    if (idx >= B_ * H_) return;
    int b = idx >> 10, h = idx & (H_ - 1);
    float acc = 0.0f;
    for (int l = 0; l < S_; ++l)
        acc += aw[(size_t)b * S_ + l] * (float)enc_out_b[((size_t)l * B_ + b) * H_ + h];
    ctx_b[idx] = (bf16_t)acc;
}

// ------------- log_softmax over 1024 cols; 1 block (256 thr) per row -------------
__global__ void log_softmax1024(const float* __restrict__ in, float* __restrict__ out)
{
    __shared__ float red[256];
    int b = blockIdx.x, t = threadIdx.x;
    const float* r = in + (size_t)b * O_;
    float mx = -INFINITY;
    for (int j = t; j < O_; j += 256) mx = fmaxf(mx, r[j]);
    red[t] = mx; __syncthreads();
    for (int s = 128; s > 0; s >>= 1) { if (t < s) red[t] = fmaxf(red[t], red[t + s]); __syncthreads(); }
    mx = red[0]; __syncthreads();
    float sum = 0.0f;
    for (int j = t; j < O_; j += 256) sum += __expf(r[j] - mx);
    red[t] = sum; __syncthreads();
    for (int s = 128; s > 0; s >>= 1) { if (t < s) red[t] += red[t + s]; __syncthreads(); }
    float lse = mx + __logf(red[0]);
    for (int j = t; j < O_; j += 256) out[(size_t)b * O_ + j] = r[j] - lse;
}

// ---------------------------------------------------------------------
static inline size_t align256(size_t x) { return (x + 255) & ~(size_t)255; }

extern "C" void kernel_launch(void* const* d_in, const int* in_sizes, int n_in,
                              void* d_out, int out_size, void* d_ws, size_t ws_size,
                              hipStream_t stream)
{
    const float* input   = (const float*)d_in[0];   // [S,B,IN]
    const float* target  = (const float*)d_in[1];   // [T,B,H]
    const float* enc_Wih = (const float*)d_in[2];
    const float* enc_Whh = (const float*)d_in[3];
    const float* enc_bih = (const float*)d_in[4];
    const float* enc_bhh = (const float*)d_in[5];
    const float* attn_W  = (const float*)d_in[6];
    const float* attn_b  = (const float*)d_in[7];
    const float* comb_W  = (const float*)d_in[8];
    const float* comb_b  = (const float*)d_in[9];
    const float* dec_Wih = (const float*)d_in[10];
    const float* dec_Whh = (const float*)d_in[11];
    const float* dec_bih = (const float*)d_in[12];
    const float* dec_bhh = (const float*)d_in[13];
    const float* out_W   = (const float*)d_in[14];
    const float* out_b   = (const float*)d_in[15];
    float* out = (float*)d_out;

    // ---------------- workspace carve-up ----------------
    char* p = (char*)d_ws;
    size_t off = 0;
    auto take = [&](size_t bytes) { void* q = p + off; off += align256(bytes); return q; };

    bf16_t* enc_Wih_b = (bf16_t*)take((size_t)H3_ * INP_ * 2);
    bf16_t* enc_Whh_b = (bf16_t*)take((size_t)H3_ * H_  * 2);
    bf16_t* dec_Wih_b = (bf16_t*)take((size_t)H3_ * H_  * 2);
    bf16_t* dec_Whh_b = (bf16_t*)take((size_t)H3_ * H_  * 2);
    bf16_t* comb_W_b  = (bf16_t*)take((size_t)H_  * 2 * H_ * 2);   // [H, 2H]
    bf16_t* attn_W_b  = (bf16_t*)take((size_t)S_  * 2 * H_ * 2);   // [S, 2H]
    bf16_t* out_W_b   = (bf16_t*)take((size_t)O_  * H_  * 2);
    bf16_t* x_b       = (bf16_t*)take((size_t)S_ * B_ * INP_ * 2);
    bf16_t* tgt_b     = (bf16_t*)take((size_t)T_ * B_ * H_   * 2);
    bf16_t* enc_out_b = (bf16_t*)take((size_t)S_ * B_ * H_ * 2);   // bf16: halves ctx stream
    float*  h         = (float*) take((size_t)B_ * H_ * 4);
    bf16_t* hb        = (bf16_t*)take((size_t)B_ * H_ * 2);
    float*  gi        = (float*) take((size_t)B_ * H3_ * 4);
    float*  gh        = (float*) take((size_t)B_ * H3_ * 4);
    float*  attn_lg   = (float*) take((size_t)B_ * S_ * 4);
    float*  aw        = (float*) take((size_t)B_ * S_ * 4);
    bf16_t* ctx_b     = (bf16_t*)take((size_t)B_ * H_ * 2);
    bf16_t* comb_bf   = (bf16_t*)take((size_t)B_ * H_ * 2);
    float*  logits    = (float*) take((size_t)B_ * O_ * 4);
    (void)ws_size; (void)in_sizes; (void)n_in; (void)out_size;

    // ---------------- one-time bf16 conversion (weights + sequences) ----------------
    auto cvt = [&](const float* s, bf16_t* d, long long rows, int sc, int dc) {
        long long n = rows * dc;
        cvt_pad_bf16<<<(unsigned)((n + 255) / 256), 256, 0, stream>>>(s, d, rows, sc, dc);
    };
    cvt(enc_Wih, enc_Wih_b, H3_, IN_, INP_);
    cvt(enc_Whh, enc_Whh_b, H3_, H_, H_);
    cvt(dec_Wih, dec_Wih_b, H3_, H_, H_);
    cvt(dec_Whh, dec_Whh_b, H3_, H_, H_);
    cvt(comb_W,  comb_W_b,  H_,  2 * H_, 2 * H_);
    cvt(attn_W,  attn_W_b,  S_,  2 * H_, 2 * H_);
    cvt(out_W,   out_W_b,   O_,  H_, H_);
    cvt(input,   x_b,   (long long)S_ * B_, IN_, INP_);
    cvt(target,  tgt_b, (long long)T_ * B_, H_,  H_);

    init_h<<<(B_ * H_ + 255) / 256, 256, 0, stream>>>(h, hb, B_ * H_);

    // Single-segment GEMM: main config (4 m-rows x 2 n-cols, 128x128/block)
    auto gemm1 = [&](const bf16_t* A, int lda, const bf16_t* W, int K,
                     const float* bias, float* Cf, bf16_t* Cb, int N, int act) {
        gemm_bf16_wmma<2><<<dim3(N / 128, B_ / 128), 256, 0, stream>>>(
            A, lda, W, K, K, nullptr, 0, nullptr, 0, 0, bias, Cf, Cb, N, act);
    };
    // Dual-segment GEMM ([x|y] @ W^T split at K0, W row stride = K0+K1)
    auto gemm2 = [&](const bf16_t* A0, const bf16_t* A1, const bf16_t* W,
                     const float* bias, float* Cf, bf16_t* Cb, int N, int act) {
        gemm_bf16_wmma<2><<<dim3(N / 128, B_ / 128), 256, 0, stream>>>(
            A0, H_, W, 2 * H_, H_, A1, H_, W + H_, 2 * H_, H_, bias, Cf, Cb, N, act);
    };
    const int nPW = (B_ * H_ + 255) / 256;

    // ---------------- encoder: 64 sequential GRU steps ----------------
    for (int t = 0; t < S_; ++t) {
        gemm1(x_b + (size_t)t * B_ * INP_, INP_, enc_Wih_b, INP_, enc_bih, gi, nullptr, H3_, 0);
        gemm1(hb, H_, enc_Whh_b, H_, enc_bhh, gh, nullptr, H3_, 0);
        gru_pointwise<<<nPW, 256, 0, stream>>>(gi, gh, h, hb, enc_out_b + (size_t)t * B_ * H_);
    }

    // ---------------- decoder: 31 sequential attention steps ----------------
    for (int i = 0; i < T_ - 1; ++i) {
        const bf16_t* x = tgt_b + (size_t)i * B_ * H_;
        // attn logits = [x|h] @ attn_W^T + attn_b  (N=64: 8x1 wave layout, 256x64/block)
        gemm_bf16_wmma<1><<<dim3(1, B_ / 256), 256, 0, stream>>>(
            x, H_, attn_W_b, 2 * H_, H_, hb, H_, attn_W_b + H_, 2 * H_, H_,
            attn_b, attn_lg, nullptr, S_, 0);
        softmax64<<<B_, 64, 0, stream>>>(attn_lg, aw);
        ctx_einsum<<<nPW, 256, 0, stream>>>(aw, enc_out_b, ctx_b);
        // comb = relu([x|ctx] @ comb_W^T + comb_b), bf16 out
        gemm2(x, ctx_b, comb_W_b, comb_b, nullptr, comb_bf, H_, 1);
        gemm1(comb_bf, H_, dec_Wih_b, H_, dec_bih, gi, nullptr, H3_, 0);
        gemm1(hb,      H_, dec_Whh_b, H_, dec_bhh, gh, nullptr, H3_, 0);
        gru_pointwise<<<nPW, 256, 0, stream>>>(gi, gh, h, hb, nullptr);
    }

    // ---------------- only last step's output survives in the reference ----------------
    gemm1(hb, H_, out_W_b, H_, out_b, logits, nullptr, O_, 0);
    log_softmax1024<<<B_, 256, 0, stream>>>(logits, out);
}
